// TritonGPTQLinear_44126493999232
// MI455X (gfx1250) — matrix operand order repaired and checked by reading, not compiled, not run on hardware
//
#include <hip/hip_runtime.h>

typedef _Float16 half2_t __attribute__((ext_vector_type(2)));
typedef _Float16 half8_t __attribute__((ext_vector_type(8)));
typedef _Float16 v16h    __attribute__((ext_vector_type(16)));
typedef float    v8f     __attribute__((ext_vector_type(8)));

static constexpr int Mdim = 4096;
static constexpr int Kdim = 4096;
static constexpr int Ndim = 11008;
static constexpr int BM = 128, BN = 128, BK = 64;
static constexpr int APAD = BK + 8;   // halves per As row
static constexpr int BPAD = BK + 8;   // halves per Bt row

union frag_u { v16h v; half8_t p[2]; };

__global__ __launch_bounds__(256, 2)
void gptq_wmma_kernel(const _Float16* __restrict__ x,
                      const unsigned int* __restrict__ qweight,
                      const _Float16* __restrict__ scales,
                      const unsigned int* __restrict__ qzeros,
                      const int* __restrict__ g_idx,
                      _Float16* __restrict__ out)
{
    __shared__ __align__(16) _Float16 As[BM * APAD];  // A tile, row-major [m][k]
    __shared__ __align__(16) _Float16 Bt[BN * BPAD];  // dequant W tile, [n][k]

    const int tid   = threadIdx.x;
    const int wave  = tid >> 5;        // 0..7
    const int lane  = tid & 31;        // wave32
    const int waveM = wave >> 2;       // 0..1  (64 rows each)
    const int waveN = wave & 3;        // 0..3  (32 cols each)
    const int lhalf = lane & 15;
    const int lhi   = lane >> 4;       // 0 or 1

    const int n0 = blockIdx.x * BN;
    const int m0 = blockIdx.y * BM;

    v8f acc[4][2];
    const v8f zero = {};
    #pragma unroll
    for (int i = 0; i < 4; ++i)
        #pragma unroll
        for (int j = 0; j < 2; ++j)
            acc[i][j] = zero;

    for (int k0 = 0; k0 < Kdim; k0 += BK) {
        // ---- stage A: 128x64 fp16 tile via b128 copies (4 per thread) ----
        #pragma unroll
        for (int it = 0; it < 4; ++it) {
            int i   = tid + it * 256;          // 0..1023
            int row = i >> 3;                  // 0..127
            int cw  = (i & 7) * 8;             // half-offset within row: 0..56
            const uint4* src = reinterpret_cast<const uint4*>(
                x + (size_t)(m0 + row) * Kdim + k0 + cw);
            *reinterpret_cast<uint4*>(&As[row * APAD + cw]) = *src;
        }

        // ---- stage B: dequant int4 64x128 chunk -> Bt[n][k] fp16 ----
        // Magic-number dequant: 0x6400|q == fp16(1024+q) exactly; subtract the
        // exact constant (1024+z) via v_pk_add_f16, scale via v_pk_mul_f16.
        // Single rounding on the multiply -> bit-identical to fp16 (q-z)*s.
        #pragma unroll
        for (int it = 0; it < 4; ++it) {
            int i  = tid + it * 256;           // 0..1023
            int n  = i & 127;                  // column within tile (coalesced)
            int kq = i >> 7;                   // 0..7 -> k_local = kq*8 .. +8
            int gc = n0 + n;                   // global column
            unsigned int w = qweight[(size_t)(k0 / 8 + kq) * Ndim + gc];
            int g = g_idx[k0 + kq * 8];        // group (constant across the 8 k's)
            _Float16 s = scales[(size_t)g * Ndim + gc];
            unsigned int zp = qzeros[(size_t)g * (Ndim / 8) + (gc >> 3)];
            int z = (int)((zp >> ((gc & 7) * 4)) & 15u) + 1;   // 1..16
            unsigned int nb  = 0xE400u + (unsigned int)z;      // bits of -(1024+z)
            half2_t negc = __builtin_bit_cast(half2_t, nb | (nb << 16));
            half2_t s2; s2[0] = s; s2[1] = s;
            half8_t v;
            #pragma unroll
            for (int j = 0; j < 4; ++j) {
                unsigned int t = w >> (8 * j);
                unsigned int p = (t & 0xFu) | ((t & 0xF0u) << 12) | 0x64006400u;
                half2_t h = __builtin_bit_cast(half2_t, p);    // {1024+q_lo,1024+q_hi}
                half2_t d = (h + negc) * s2;                   // pk_add + pk_mul
                v[2 * j]     = d[0];
                v[2 * j + 1] = d[1];
            }
            *reinterpret_cast<half8_t*>(&Bt[n * BPAD + kq * 8]) = v;
        }
        __syncthreads();

        // ---- two WMMA K-substeps of 32 ----
        #pragma unroll
        for (int ks = 0; ks < BK; ks += 32) {
            v16h afrag[4], bfrag[2];
            #pragma unroll
            for (int mt = 0; mt < 4; ++mt) {
                // A 16x32 f16: lanes 0-15 row M=lane: elems 0..7=K0..7, 8..15=K16..23
                //              lanes 16-31 same rows, K+8
                int rb = (waveM * 64 + mt * 16 + lhalf) * APAD + ks;
                frag_u u;
                u.p[0] = *reinterpret_cast<const half8_t*>(&As[rb + lhi * 8]);
                u.p[1] = *reinterpret_cast<const half8_t*>(&As[rb + 16 + lhi * 8]);
                afrag[mt] = u.v;
            }
            #pragma unroll
            for (int nt = 0; nt < 2; ++nt) {
                // B 32x16 f16: lanes 0-15 col N=lane: elems 0..7=K0..7, 8..15=K16..23; lanes 16-31 K+8
                int rb = (waveN * 32 + nt * 16 + lhalf) * BPAD + ks + lhi * 8;
                frag_u u;
                u.p[0] = *reinterpret_cast<const half8_t*>(&Bt[rb]);
                u.p[1] = *reinterpret_cast<const half8_t*>(&Bt[rb + 16]);
                bfrag[nt] = u.v;
            }
            #pragma unroll
            for (int mt = 0; mt < 4; ++mt)
                #pragma unroll
                for (int nt = 0; nt < 2; ++nt)
                    acc[mt][nt] = __builtin_amdgcn_wmma_f32_16x16x32_f16(
                        false, afrag[mt], false, bfrag[nt],
                        (short)0, acc[mt][nt], false, false);
        }
        __syncthreads();
    }

    // ---- epilogue: C/D layout -> fp16 stores ----
    // VGPR r: lanes 0-15 hold (M=r, N=lane); lanes 16-31 hold (M=r+8, N=lane-16)
    #pragma unroll
    for (int mt = 0; mt < 4; ++mt) {
        #pragma unroll
        for (int nt = 0; nt < 2; ++nt) {
            int ng = n0 + waveN * 32 + nt * 16 + lhalf;
            int mb = m0 + waveM * 64 + mt * 16 + lhi * 8;
            #pragma unroll
            for (int r = 0; r < 8; ++r)
                out[(size_t)(mb + r) * Ndim + ng] = (_Float16)acc[mt][nt][r];
        }
    }
}

extern "C" void kernel_launch(void* const* d_in, const int* in_sizes, int n_in,
                              void* d_out, int out_size, void* d_ws, size_t ws_size,
                              hipStream_t stream) {
    (void)in_sizes; (void)n_in; (void)out_size; (void)d_ws; (void)ws_size;
    const _Float16*     x       = (const _Float16*)d_in[0];
    const unsigned int* qweight = (const unsigned int*)d_in[1];
    const _Float16*     scales  = (const _Float16*)d_in[2];
    const unsigned int* qzeros  = (const unsigned int*)d_in[3];
    const int*          g_idx   = (const int*)d_in[4];
    _Float16*           out     = (_Float16*)d_out;

    dim3 grid(Ndim / BN, Mdim / BM);   // 86 x 32 workgroups
    dim3 block(256);                   // 8 waves (wave32)
    gptq_wmma_kernel<<<grid, block, 0, stream>>>(x, qweight, scales, qzeros, g_idx, out);
}
